// GGNN_65120294142423
// MI455X (gfx1250) — matrix operand order repaired and checked by reading, not compiled, not run on hardware
//
#include <hip/hip_runtime.h>
#include <hip/hip_bf16.h>
#include <math.h>

// ---------------------------------------------------------------------------
// GatedGraphConv x2 for MI455X (gfx1250, wave32, WMMA bf16)
//   layer: m = adj @ (x@W); h = relu(GRUCell(m, x))
// Dominant op: adj[12288x12288] @ B[12288x64]  -> v_wmma_f32_16x16x32_bf16
//   - adj converted fp32->bf16 in-register (1 v_perm_b32 per pair)
//   - B pre-swizzled into WMMA fragment order (no LDS, no transpose)
//   - K-split x4 + fixed-order reduce for occupancy (deterministic)
//   - manual double-buffering of A and B fragment loads
// ---------------------------------------------------------------------------

#define NN 12288
#define DD 64

typedef __attribute__((ext_vector_type(16))) __bf16   v16bf;
typedef __attribute__((ext_vector_type(8)))  float    v8f;
typedef __attribute__((ext_vector_type(4)))  float    vf4;
typedef __attribute__((ext_vector_type(4)))  unsigned vu4;

// ---- fp32 -> bf16 helpers --------------------------------------------------
static __device__ __forceinline__ unsigned short bf16_rne(float f) {
  union { float f; unsigned u; } v; v.f = f;
  unsigned r = (v.u + 0x7FFFu + ((v.u >> 16) & 1u)) >> 16;
  return (unsigned short)r;
}

// pack two fp32 into {bf16(b), bf16(a)}.
static __device__ __forceinline__ unsigned pk_bf16(float a, float b) {
#if __has_builtin(__builtin_amdgcn_cvt_pk_bf16_f32)
  typedef __attribute__((ext_vector_type(2))) __bf16 v2bf;
  v2bf t = __builtin_amdgcn_cvt_pk_bf16_f32(a, b);
  unsigned r; __builtin_memcpy(&r, &t, 4);
  return r;
#else
  // single v_perm_b32: result bytes = {b[3], b[2], a[3], a[2]}  (truncation)
  union { float f; unsigned u; } ua, ub; ua.f = a; ub.f = b;
  return __builtin_amdgcn_perm(ub.u, ua.u, 0x07060302u);
#endif
}

// ---------------------------------------------------------------------------
// Kernel A: xw = X @ W  (fp32 in, bf16 out), stored pre-swizzled in the
// contiguous-k B-fragment order (same k order as the A packing):
//   32x16 bf16 tile: lane L holds N = L%16, K = (L/16)*16 + i, i = 0..15
//   (contiguous 32 bytes per lane).  frag f = (k/32)*4 + (col/16);
//   ushort addr = f*512 + L*16 + i.
// ---------------------------------------------------------------------------
__global__ __launch_bounds__(256) void xw_bf16_swizzle(
    const float* __restrict__ X,   // [NN, DD]
    const float* __restrict__ W,   // [DD, DD]
    unsigned short* __restrict__ bsw) {
  __shared__ float Wl[DD * DD];
  for (int t = threadIdx.x; t < DD * DD; t += 256) Wl[t] = W[t];
  __syncthreads();

  const int g   = blockIdx.x * 256 + threadIdx.x;
  const int k   = g >> 6;     // row of X == K index of B
  const int col = g & 63;

  const float* xr = X + (size_t)k * DD;
  float acc = 0.f;
#pragma unroll
  for (int t = 0; t < DD; ++t) acc = fmaf(xr[t], Wl[t * DD + col], acc);

  const int kt    = k >> 5;
  const int r5    = k & 31;
  const int khalf = r5 >> 4;
  const int i     = r5 & 15;
  const int lane  = khalf * 16 + (col & 15);
  const int frag  = kt * 4 + (col >> 4);
  bsw[(size_t)frag * 512 + lane * 16 + i] = bf16_rne(acc);
}

// ---------------------------------------------------------------------------
// Kernel B: partial m = adj[:, kBeg:kEnd] @ B[kBeg:kEnd, :]
// Block = 256 threads = 8 waves; each wave: 16 rows x 64 cols (4 v8f accs).
// grid = (NN/128, S);  blockIdx.y selects the K slice + output slice.
// Double-buffered A (nontemporal fp32) and B (bf16 frags, L2 resident).
// ---------------------------------------------------------------------------
__global__ __launch_bounds__(256) void ggc_adj_gemm(
    const float* __restrict__ adj,       // [NN, NN]
    const unsigned* __restrict__ bsw,    // swizzled bf16 fragments
    float* __restrict__ mSlices,         // [S, NN, DD]
    int kChunk) {                        // NN / S, multiple of 32
  const int lane = threadIdx.x & 31;
  const int wave = threadIdx.x >> 5;
  const int half = lane >> 4;
  const int rlan = lane & 15;
  const int row0 = blockIdx.x * 128 + wave * 16;
  const int row  = row0 + rlan;

  const int kBeg = blockIdx.y * kChunk;
  const int kEnd = kBeg + kChunk;
  float* __restrict__ m = mSlices + (size_t)blockIdx.y * NN * DD;

  v8f acc0 = {}, acc1 = {}, acc2 = {}, acc3 = {};
  const float* arow = adj + (size_t)row * NN;
  const vf4*   bq4  = (const vf4*)arow;          // unused; silence none
  (void)bq4;
  const vu4* bq = (const vu4*)bsw;

  // ---- prologue: load tile kBeg -------------------------------------------
  const float* ap0 = arow + kBeg + half * 16;
  vf4 a0 = __builtin_nontemporal_load((const vf4*)ap0);
  vf4 a1 = __builtin_nontemporal_load((const vf4*)ap0 + 1);
  vf4 a2 = __builtin_nontemporal_load((const vf4*)(ap0 + 8));
  vf4 a3 = __builtin_nontemporal_load((const vf4*)(ap0 + 8) + 1);
  int fb = ((kBeg >> 5) * 4) * 64 + lane * 2;    // vu4 index of frag base
  vu4 b0 = bq[fb +   0], b1 = bq[fb +   1];
  vu4 b2 = bq[fb +  64], b3 = bq[fb +  65];
  vu4 b4 = bq[fb + 128], b5 = bq[fb + 129];
  vu4 b6 = bq[fb + 192], b7 = bq[fb + 193];

#pragma unroll 2
  for (int kt = kBeg; kt < kEnd; kt += 32) {
    // ---- issue next tile's loads (double buffer) --------------------------
    const int ktn = (kt + 32 < kEnd) ? kt + 32 : kBeg;   // harmless wrap
    const float* apn = arow + ktn + half * 16;
    vf4 n0 = __builtin_nontemporal_load((const vf4*)apn);
    vf4 n1 = __builtin_nontemporal_load((const vf4*)apn + 1);
    vf4 n2 = __builtin_nontemporal_load((const vf4*)(apn + 8));
    vf4 n3 = __builtin_nontemporal_load((const vf4*)(apn + 8) + 1);
    const int fbn = ((ktn >> 5) * 4) * 64 + lane * 2;
    vu4 nb0 = bq[fbn +   0], nb1 = bq[fbn +   1];
    vu4 nb2 = bq[fbn +  64], nb3 = bq[fbn +  65];
    vu4 nb4 = bq[fbn + 128], nb5 = bq[fbn + 129];
    vu4 nb6 = bq[fbn + 192], nb7 = bq[fbn + 193];

    // ---- pack current A fragment (16x32 bf16, contiguous k = 16*half + s) ---
    union { unsigned u[8]; v16bf v; } A;
    A.u[0] = pk_bf16(a0.x, a0.y); A.u[1] = pk_bf16(a0.z, a0.w);
    A.u[2] = pk_bf16(a1.x, a1.y); A.u[3] = pk_bf16(a1.z, a1.w);
    A.u[4] = pk_bf16(a2.x, a2.y); A.u[5] = pk_bf16(a2.z, a2.w);
    A.u[6] = pk_bf16(a3.x, a3.y); A.u[7] = pk_bf16(a3.z, a3.w);

    union { vu4 q[2]; v16bf v; } B0, B1, B2, B3;
    B0.q[0] = b0; B0.q[1] = b1;
    B1.q[0] = b2; B1.q[1] = b3;
    B2.q[0] = b4; B2.q[1] = b5;
    B3.q[0] = b6; B3.q[1] = b7;

    acc0 = __builtin_amdgcn_wmma_f32_16x16x32_bf16(false, A.v, false, B0.v,
                                                   (short)0, acc0, false, false);
    acc1 = __builtin_amdgcn_wmma_f32_16x16x32_bf16(false, A.v, false, B1.v,
                                                   (short)0, acc1, false, false);
    acc2 = __builtin_amdgcn_wmma_f32_16x16x32_bf16(false, A.v, false, B2.v,
                                                   (short)0, acc2, false, false);
    acc3 = __builtin_amdgcn_wmma_f32_16x16x32_bf16(false, A.v, false, B3.v,
                                                   (short)0, acc3, false, false);

    // ---- rotate buffers ---------------------------------------------------
    a0 = n0; a1 = n1; a2 = n2; a3 = n3;
    b0 = nb0; b1 = nb1; b2 = nb2; b3 = nb3;
    b4 = nb4; b5 = nb5; b6 = nb6; b7 = nb7;
  }

  // ---- store: 32-bit C/D layout: VGPR j -> M = half*8 + j, N = rlan -------
#pragma unroll
  for (int j = 0; j < 8; ++j) {
    const size_t r = (size_t)(row0 + half * 8 + j) * DD;
    m[r +  0 + rlan] = acc0[j];
    m[r + 16 + rlan] = acc1[j];
    m[r + 32 + rlan] = acc2[j];
    m[r + 48 + rlan] = acc3[j];
  }
}

// ---------------------------------------------------------------------------
// Kernel R: deterministic fixed-order reduction of K-split partials into
// slice 0.  m[0] += m[1] + m[2] + ... (in index order).
// ---------------------------------------------------------------------------
__global__ __launch_bounds__(256) void reduce_slices(
    float* __restrict__ m, int S) {
  const size_t g = (size_t)blockIdx.x * 256 + threadIdx.x;
  float a = m[g];
  for (int s = 1; s < S; ++s) a += m[(size_t)s * NN * DD + g];
  m[g] = a;
}

// ---------------------------------------------------------------------------
// Kernel C: fused GRU cell + ReLU, one thread per (row, col).
// ---------------------------------------------------------------------------
__global__ __launch_bounds__(256) void gru_relu(
    const float* __restrict__ M,    // [NN, DD] (summed)
    const float* __restrict__ Xin,  // [NN, DD]
    const float* __restrict__ wih,  // [3*DD, DD]
    const float* __restrict__ whh,  // [3*DD, DD]
    const float* __restrict__ bih,  // [3*DD]
    const float* __restrict__ bhh,  // [3*DD]
    float* __restrict__ out) {      // [NN, DD]
  const int g   = blockIdx.x * 256 + threadIdx.x;
  const int row = g >> 6;
  const int col = g & 63;

  const float* mr  = M   + (size_t)row * DD;
  const float* xr  = Xin + (size_t)row * DD;
  const float* wr  = wih + (size_t)col * DD;   // rows col, col+64, col+128
  const float* hr_ = whh + (size_t)col * DD;

  float ir = bih[col], iz = bih[col + 64], in_ = bih[col + 128];
  float hr = bhh[col], hz = bhh[col + 64], hn  = bhh[col + 128];

#pragma unroll 4
  for (int k = 0; k < DD; ++k) {
    const float mk = mr[k], xk = xr[k];
    ir  = fmaf(mk, wr [k],            ir);
    iz  = fmaf(mk, wr [64 * DD + k],  iz);
    in_ = fmaf(mk, wr [128 * DD + k], in_);
    hr  = fmaf(xk, hr_[k],            hr);
    hz  = fmaf(xk, hr_[64 * DD + k],  hz);
    hn  = fmaf(xk, hr_[128 * DD + k], hn);
  }

  const float r = 1.f / (1.f + __expf(-(ir + hr)));
  const float z = 1.f / (1.f + __expf(-(iz + hz)));
  const float n = tanhf(in_ + r * hn);
  const float h = (1.f - z) * n + z * xr[col];
  out[g] = fmaxf(h, 0.f);
}

// ---------------------------------------------------------------------------
extern "C" void kernel_launch(void* const* d_in, const int* in_sizes, int n_in,
                              void* d_out, int out_size, void* d_ws, size_t ws_size,
                              hipStream_t stream) {
  const float* x    = (const float*)d_in[0];
  const float* adj  = (const float*)d_in[1];
  const float* W1   = (const float*)d_in[2];
  const float* wih1 = (const float*)d_in[3];
  const float* whh1 = (const float*)d_in[4];
  const float* bih1 = (const float*)d_in[5];
  const float* bhh1 = (const float*)d_in[6];
  const float* W2   = (const float*)d_in[7];
  const float* wih2 = (const float*)d_in[8];
  const float* whh2 = (const float*)d_in[9];
  const float* bih2 = (const float*)d_in[10];
  const float* bhh2 = (const float*)d_in[11];

  // K-split factor: 4 if workspace allows (17 MB), else 1 (deterministic
  // for a given harness configuration).
  const size_t MB = 1024u * 1024u;
  const int S = (ws_size >= 17 * MB) ? 4 : 1;

  char* ws = (char*)d_ws;
  unsigned short* bsw = (unsigned short*)ws;             // 1.5 MB @ 0
  float* m = (float*)(ws + 2 * MB);                      // S x 3 MB @ 2 MB
  float* h = (float*)(ws + (2 + 3 * (size_t)S) * MB);    // 3 MB after slices
  float* out = (float*)d_out;

  const dim3 blkE(256), grdE(NN * DD / 256);   // 3072 blocks
  const dim3 blkG(256);
  const dim3 grdG(NN / 128, S);                // 96 x S blocks, 8 waves each
  const int  kChunk = NN / S;

  // ---- layer 1 ----
  xw_bf16_swizzle<<<grdE, blkE, 0, stream>>>(x, W1, bsw);
  ggc_adj_gemm  <<<grdG, blkG, 0, stream>>>(adj, (const unsigned*)bsw, m, kChunk);
  if (S > 1) reduce_slices<<<grdE, blkE, 0, stream>>>(m, S);
  gru_relu      <<<grdE, blkE, 0, stream>>>(m, x, wih1, whh1, bih1, bhh1, h);

  // ---- layer 2 ----
  xw_bf16_swizzle<<<grdE, blkE, 0, stream>>>(h, W2, bsw);
  ggc_adj_gemm  <<<grdG, blkG, 0, stream>>>(adj, (const unsigned*)bsw, m, kChunk);
  if (S > 1) reduce_slices<<<grdE, blkE, 0, stream>>>(m, S);
  gru_relu      <<<grdE, blkE, 0, stream>>>(m, h, wih2, whh2, bih2, bhh2, out);

  (void)in_sizes; (void)n_in; (void)out_size;
}
